// AdaConvBlock_10307921510661
// MI455X (gfx1250) — matrix operand order repaired and checked by reading, not compiled, not run on hardware
//
#include <hip/hip_runtime.h>
#include <math.h>

// ---------------- problem constants ----------------
#define BB   16
#define CC   384
#define LL   4096
#define CT   128        // C/3
#define SIXC 2304       // 6*C
#define KLEN 1024

typedef unsigned short u16;
typedef __attribute__((ext_vector_type(16))) __bf16 v16bf;
typedef __attribute__((ext_vector_type(8)))  float  v8f;

union ABfrag { v16bf v; uint4 q[2]; };

__device__ __forceinline__ u16 f2bf(float f) {
    unsigned u = __float_as_uint(f);
    unsigned r = u + 0x7FFFu + ((u >> 16) & 1u);   // RNE
    return (u16)(r >> 16);
}
__device__ __forceinline__ float bf2f(u16 s) {
    return __uint_as_float(((unsigned)s) << 16);
}
__device__ __forceinline__ v8f wmma_bf16(const ABfrag& a, const ABfrag& b, v8f c) {
    // D = A(16x32 bf16) * B(32x16 bf16) + C(16x16 f32)
    return __builtin_amdgcn_wmma_f32_16x16x32_bf16(false, a.v, false, b.v,
                                                   (short)0, c, false, false);
}

// ---------------- workspace layout (bytes) ----------------
static constexpr size_t OFF_KFULL = 0;                                   // 384*1024 f32
static constexpr size_t OFF_ADAW  = 1572864;                             // 2304*128 bf16
static constexpr size_t OFF_W1B   = OFF_ADAW + 589824;                   // 384*384 bf16
static constexpr size_t OFF_W2B   = OFF_W1B  + 294912;                   // 384*384 bf16
static constexpr size_t OFF_MOD   = OFF_W2B  + 294912;                   // B*L*2304 bf16 (b,l,6C)
static constexpr size_t OFF_U     = OFF_MOD  + (size_t)BB * LL * SIXC * 2; // B*C*L f32 (b,c,l)
static constexpr size_t WS_NEED   = OFF_U    + (size_t)BB * CC * LL * 4;

// ================= K0a: fp32 -> bf16 weight convert =================
__global__ void k_f2bf(const float* __restrict__ src, u16* __restrict__ dst, int n) {
    int i = blockIdx.x * 256 + threadIdx.x;
    if (i < n) dst[i] = f2bf(src[i]);
}

// ================= K0b: build normalized 1024-tap kernel per channel =================
__global__ void k_buildk(const float* __restrict__ kern, float* __restrict__ kfull) {
    const int c = blockIdx.x;
    const int tid = threadIdx.x;
    __shared__ float red[256];
    float vals[4];
    float ss = 0.f;
#pragma unroll
    for (int e = 0; e < 4; ++e) {
        int t = tid * 4 + e;             // 0..1023
        int s, j;
        if      (t <  32) { s = 0; j = t; }
        else if (t <  64) { s = 1; j = t - 32; }
        else if (t < 128) { s = 2; j = (t - 64)  >> 1; }
        else if (t < 256) { s = 3; j = (t - 128) >> 2; }
        else if (t < 512) { s = 4; j = (t - 256) >> 3; }
        else              { s = 5; j = (t - 512) >> 4; }
        float scale = (float)(1 << (5 - s));            // DECAY^(NS-1-s), DECAY=2
        float v = kern[((size_t)s * CC + c) * 32 + j] * scale;
        vals[e] = v;
        ss += v * v;
    }
    red[tid] = ss; __syncthreads();
    for (int off = 128; off > 0; off >>= 1) {
        if (tid < off) red[tid] += red[tid + off];
        __syncthreads();
    }
    float rn = rsqrtf(red[0]);
#pragma unroll
    for (int e = 0; e < 4; ++e)
        kfull[(size_t)c * KLEN + tid * 4 + e] = vals[e] * rn;
}

// ================= K1: ada modulation GEMM (WMMA bf16) =================
// temp(b,l,o) = silu(t_cond(b,:,l)) . ada_w(o,:) + ada_b(o), stored bf16 as (b,l,2304)
// Block: 16 positions x 384 outputs; 8 waves x 3 N-tiles/wave; K-outer so A is loaded once.
__global__ void k_ada(const float* __restrict__ tcond, const u16* __restrict__ adaW,
                      const float* __restrict__ adab, u16* __restrict__ mod) {
    const int b  = blockIdx.x >> 8;
    const int l0 = (blockIdx.x & 255) * 16;
    const int tid = threadIdx.x;

    __shared__ __align__(16) u16 tl[16 * CT];      // [pos][k], silu applied
    for (int idx = tid; idx < 16 * CT; idx += 256) {
        int p = idx & 15, k = idx >> 4;
        float v = tcond[((size_t)b * CT + k) * LL + l0 + p];
        v = v / (1.f + __expf(-v));                // silu
        tl[p * CT + k] = f2bf(v);
    }
    __syncthreads();

    const int lane = tid & 31, w = tid >> 5;
    const int hi = lane >> 4, ln = lane & 15;

    const u16* brow[3];
    int orow[3];
#pragma unroll
    for (int jt = 0; jt < 3; ++jt) {
        orow[jt] = blockIdx.y * 384 + (w + 8 * jt) * 16 + ln;
        brow[jt] = adaW + (size_t)orow[jt] * CT;
    }

    v8f acc[3] = {{0.f,0.f,0.f,0.f,0.f,0.f,0.f,0.f},
                  {0.f,0.f,0.f,0.f,0.f,0.f,0.f,0.f},
                  {0.f,0.f,0.f,0.f,0.f,0.f,0.f,0.f}};
#pragma unroll
    for (int kk = 0; kk < 4; ++kk) {
        const int kb = kk * 32;
        ABfrag a;
        a.q[0] = *(const uint4*)(tl + ln * CT + kb + hi * 8);
        a.q[1] = *(const uint4*)(tl + ln * CT + kb + 16 + hi * 8);
#pragma unroll
        for (int jt = 0; jt < 3; ++jt) {
            ABfrag bm;
            const uint4* bp = (const uint4*)(brow[jt] + kb + hi * 8);
            bm.q[0] = bp[0]; bm.q[1] = bp[2];
            acc[jt] = wmma_bf16(a, bm, acc[jt]);
        }
    }
#pragma unroll
    for (int jt = 0; jt < 3; ++jt) {
        const float bias = adab[orow[jt]];
#pragma unroll
        for (int r = 0; r < 8; ++r) {
            int l = l0 + r + hi * 8;               // D: VGPR r -> M = r (+8 for hi lanes)
            mod[((size_t)b * LL + l) * SIXC + orow[jt]] = f2bf(acc[jt][r] + bias);
        }
    }
}

// ================= K2: channel-LN + (1+scale_tm)*y + shift_tm -> ubuf =================
__global__ void k_lnmod(const float* __restrict__ x, const u16* __restrict__ mod,
                        float* __restrict__ u) {
    const int b  = blockIdx.x >> 8;
    const int l0 = (blockIdx.x & 255) * 16;
    const int tid = threadIdx.x;
    const int p = tid >> 4, q = tid & 15;

    __shared__ float r0[256], r1[256];
    float s = 0.f, s2 = 0.f;
    for (int c = q; c < CC; c += 16) {
        float v = x[((size_t)b * CC + c) * LL + l0 + p];
        s += v; s2 += v * v;
    }
    r0[tid] = s; r1[tid] = s2; __syncthreads();
    for (int off = 8; off > 0; off >>= 1) {
        if (q < off) { r0[tid] += r0[tid + off]; r1[tid] += r1[tid + off]; }
        __syncthreads();
    }
    const float mu = r0[p * 16] * (1.f / CC);
    const float var = r1[p * 16] * (1.f / CC) - mu * mu;
    const float rs = rsqrtf(var + 1e-5f);
    const size_t mbase = ((size_t)b * LL + l0 + p) * SIXC;
    for (int c = q; c < CC; c += 16) {
        float v = (x[((size_t)b * CC + c) * LL + l0 + p] - mu) * rs;
        float sc = bf2f(mod[mbase + CC + c]);      // scale_tm (group 1)
        float sh = bf2f(mod[mbase + c]);           // shift_tm (group 0)
        u[((size_t)b * CC + c) * LL + l0 + p] = v * (1.f + sc) + sh;
    }
}

// ================= K3: structured long conv via prefix sums + residual =================
// Taps repeated r times == single tap on r-wide box sum == P[a]-P[a-r] from a prefix sum.
// 192 effective taps instead of 1024.  x2 = x + gate_tm*(conv + u*D)  -> d_out
__global__ void k_conv(const float* __restrict__ x, const float* __restrict__ u,
                       const float* __restrict__ kfull, const float* __restrict__ Dv,
                       const u16* __restrict__ mod, float* __restrict__ xout) {
    constexpr int SEG_ST[6] = {0, 32, 64, 128, 256, 512};
    constexpr int SEG_R[6]  = {1, 1, 2, 4, 8, 16};
    const int i0 = blockIdx.x * 256;
    const int c  = blockIdx.y;
    const int b  = blockIdx.z;
    const int tid = threadIdx.x;

    __shared__ float win[1280];     // u window [i0-512, i0+767], then in-place prefix
    __shared__ float partial[256];
    __shared__ float wk[192];       // 6 scales x 32 taps (normalized)

    const float* urow = u + ((size_t)b * CC + c) * LL;
    const int base = i0 - 512;
    for (int t = tid; t < 1280; t += 256) {
        int j = base + t;
        win[t] = (j >= 0 && j < LL) ? urow[j] : 0.f;
    }
    if (tid < 192) {
        int s = tid >> 5, j = tid & 31;
        wk[tid] = kfull[(size_t)c * KLEN + SEG_ST[s] + SEG_R[s] * j];
    }
    __syncthreads();

    const float uc = win[tid + 512];               // u at output position (D-term)
    // block-local inclusive prefix sum of win[1280] (5 elems/thread + Hillis-Steele)
    float loc[5]; float run = 0.f;
#pragma unroll
    for (int e = 0; e < 5; ++e) { run += win[tid * 5 + e]; loc[e] = run; }
    partial[tid] = run; __syncthreads();
    for (int off = 1; off < 256; off <<= 1) {
        float t = (tid >= off) ? partial[tid - off] : 0.f;
        __syncthreads();
        partial[tid] += t;
        __syncthreads();
    }
    const float pb = (tid == 0) ? 0.f : partial[tid - 1];
#pragma unroll
    for (int e = 0; e < 5; ++e) win[tid * 5 + e] = loc[e] + pb;
    __syncthreads();

    float acc = 0.f;
#pragma unroll
    for (int s = 0; s < 6; ++s) {
        const int r = SEG_R[s], st = SEG_ST[s];
#pragma unroll
        for (int j = 0; j < 32; ++j) {
            int a = tid + 1024 - st - r * j;
            acc = fmaf(wk[s * 32 + j], win[a] - win[a - r], acc);
        }
    }
    const int i = i0 + tid;
    const float g = bf2f(mod[((size_t)b * LL + i) * SIXC + 2 * CC + c]);   // gate_tm
    const size_t xi = ((size_t)b * CC + c) * LL + i;
    xout[xi] = x[xi] + g * (acc + uc * Dv[c]);
}

// ================= K4: LN + mod + MLP (two WMMA GEMMs) + gated residual =================
__global__ void k_mlp(const u16* __restrict__ mod,
                      const u16* __restrict__ w1b, const float* __restrict__ b1v,
                      const u16* __restrict__ w2b, const float* __restrict__ b2v,
                      float* __restrict__ xio) {
    const int b  = blockIdx.x >> 8;
    const int l0 = (blockIdx.x & 255) * 16;
    const int tid = threadIdx.x;

    __shared__ __align__(16) float xt[CC * 17];    // x2 tile [c][p], pad-17 vs bank conflicts
    __shared__ __align__(16) u16  v_lds[16 * CC];  // modulated LN,   [pos][k] bf16
    __shared__ __align__(16) u16  h_lds[16 * CC];  // gelu(h),        [pos][k] bf16
    __shared__ float r0[256], r1[256];

    // stage x2 tile (coalesced reads of d_out written by K3)
    for (int idx = tid; idx < CC * 16; idx += 256) {
        int ch = idx >> 4, p = idx & 15;
        xt[ch * 17 + p] = xio[((size_t)b * CC + ch) * LL + l0 + p];
    }
    __syncthreads();

    // channel LN + cm modulation, into v_lds
    {
        const int p = tid >> 4, q = tid & 15;
        float s = 0.f, s2 = 0.f;
        for (int ch = q; ch < CC; ch += 16) {
            float v = xt[ch * 17 + p];
            s += v; s2 += v * v;
        }
        r0[tid] = s; r1[tid] = s2; __syncthreads();
        for (int off = 8; off > 0; off >>= 1) {
            if (q < off) { r0[tid] += r0[tid + off]; r1[tid] += r1[tid + off]; }
            __syncthreads();
        }
        const float mu = r0[p * 16] * (1.f / CC);
        const float var = r1[p * 16] * (1.f / CC) - mu * mu;
        const float rs = rsqrtf(var + 1e-5f);
        const size_t mbase = ((size_t)b * LL + l0 + p) * SIXC;
        for (int ch = q; ch < CC; ch += 16) {
            float v  = (xt[ch * 17 + p] - mu) * rs;
            float sc = bf2f(mod[mbase + 4 * CC + ch]);   // scale_cm
            float sh = bf2f(mod[mbase + 3 * CC + ch]);   // shift_cm
            v_lds[p * CC + ch] = f2bf(v * (1.f + sc) + sh);
        }
    }
    __syncthreads();

    const int lane = tid & 31, w = tid >> 5;
    const int hi = lane >> 4, ln = lane & 15;
    int orow[3];
#pragma unroll
    for (int jt = 0; jt < 3; ++jt) orow[jt] = (w + 8 * jt) * 16 + ln;

    // GEMM1: h = gelu(w1 . v + b1)  -- K-outer, one A load feeds 3 WMMAs
    {
        v8f acc[3] = {{0.f,0.f,0.f,0.f,0.f,0.f,0.f,0.f},
                      {0.f,0.f,0.f,0.f,0.f,0.f,0.f,0.f},
                      {0.f,0.f,0.f,0.f,0.f,0.f,0.f,0.f}};
#pragma unroll
        for (int jt = 0; jt < 3; ++jt)
            __builtin_prefetch(w2b + (size_t)orow[jt] * CC, 0, 1); // pre-pull GEMM2 weights
#pragma unroll
        for (int kk = 0; kk < 12; ++kk) {
            const int kb = kk * 32;
            ABfrag a;
            a.q[0] = *(const uint4*)(v_lds + ln * CC + kb + hi * 8);
            a.q[1] = *(const uint4*)(v_lds + ln * CC + kb + 16 + hi * 8);
#pragma unroll
            for (int jt = 0; jt < 3; ++jt) {
                ABfrag bm;
                const uint4* bp = (const uint4*)(w1b + (size_t)orow[jt] * CC + kb + hi * 8);
                bm.q[0] = bp[0]; bm.q[1] = bp[2];
                acc[jt] = wmma_bf16(a, bm, acc[jt]);
            }
        }
#pragma unroll
        for (int jt = 0; jt < 3; ++jt) {
            const float bias = b1v[orow[jt]];
#pragma unroll
            for (int r = 0; r < 8; ++r) {
                int m = r + hi * 8;
                float hv = acc[jt][r] + bias;
                hv = 0.5f * hv * (1.f + erff(hv * 0.70710678118f));   // exact gelu
                h_lds[m * CC + orow[jt]] = f2bf(hv);
            }
        }
    }
    __syncthreads();

    // GEMM2: m = w2 . h + b2; out = x2 + gate_cm * m  (in-place on d_out)
    {
        v8f acc[3] = {{0.f,0.f,0.f,0.f,0.f,0.f,0.f,0.f},
                      {0.f,0.f,0.f,0.f,0.f,0.f,0.f,0.f},
                      {0.f,0.f,0.f,0.f,0.f,0.f,0.f,0.f}};
#pragma unroll
        for (int kk = 0; kk < 12; ++kk) {
            const int kb = kk * 32;
            ABfrag a;
            a.q[0] = *(const uint4*)(h_lds + ln * CC + kb + hi * 8);
            a.q[1] = *(const uint4*)(h_lds + ln * CC + kb + 16 + hi * 8);
#pragma unroll
            for (int jt = 0; jt < 3; ++jt) {
                ABfrag bm;
                const uint4* bp = (const uint4*)(w2b + (size_t)orow[jt] * CC + kb + hi * 8);
                bm.q[0] = bp[0]; bm.q[1] = bp[2];
                acc[jt] = wmma_bf16(a, bm, acc[jt]);
            }
        }
#pragma unroll
        for (int jt = 0; jt < 3; ++jt) {
            const float bias = b2v[orow[jt]];
#pragma unroll
            for (int r = 0; r < 8; ++r) {
                int m = r + hi * 8;
                int l = l0 + m;
                float mv = acc[jt][r] + bias;
                float g = bf2f(mod[((size_t)b * LL + l) * SIXC + 5 * CC + orow[jt]]); // gate_cm
                xio[((size_t)b * CC + orow[jt]) * LL + l] = xt[orow[jt] * 17 + m] + g * mv;
            }
        }
    }
}

// ================= launch =================
extern "C" void kernel_launch(void* const* d_in, const int* in_sizes, int n_in,
                              void* d_out, int out_size, void* d_ws, size_t ws_size,
                              hipStream_t stream) {
    (void)in_sizes; (void)n_in; (void)out_size;
    if (ws_size < WS_NEED) return;   // ~405 MB scratch required

    const float* x     = (const float*)d_in[0];
    const float* tcond = (const float*)d_in[1];
    const float* kern  = (const float*)d_in[2];
    const float* Dv    = (const float*)d_in[3];
    const float* adaw  = (const float*)d_in[4];
    const float* adab  = (const float*)d_in[5];
    const float* w1    = (const float*)d_in[6];
    const float* b1    = (const float*)d_in[7];
    const float* w2    = (const float*)d_in[8];
    const float* b2    = (const float*)d_in[9];
    float* out = (float*)d_out;

    char* ws = (char*)d_ws;
    float* kfull = (float*)(ws + OFF_KFULL);
    u16*   adaWb = (u16*)(ws + OFF_ADAW);
    u16*   w1b   = (u16*)(ws + OFF_W1B);
    u16*   w2b   = (u16*)(ws + OFF_W2B);
    u16*   modb  = (u16*)(ws + OFF_MOD);
    float* ubuf  = (float*)(ws + OFF_U);

    // one-time per-launch weight prep (tiny; stays resident in L2 afterwards)
    k_f2bf<<<(SIXC * CT + 255) / 256, 256, 0, stream>>>(adaw, adaWb, SIXC * CT);
    k_f2bf<<<(CC * CC + 255) / 256, 256, 0, stream>>>(w1, w1b, CC * CC);
    k_f2bf<<<(CC * CC + 255) / 256, 256, 0, stream>>>(w2, w2b, CC * CC);
    k_buildk<<<CC, 256, 0, stream>>>(kern, kfull);

    // ada modulation GEMM: (B*L/16) x (2304/384) blocks, 8 waves, 3 WMMA tiles/wave
    k_ada<<<dim3(BB * (LL / 16), SIXC / 384), 256, 0, stream>>>(tcond, adaWb, adab, modb);

    // LN + tm-modulate -> conv input
    k_lnmod<<<BB * (LL / 16), 256, 0, stream>>>(x, modb, ubuf);

    // structured long conv + gated residual -> d_out holds x2
    k_conv<<<dim3(LL / 256, CC, BB), 256, 0, stream>>>(x, ubuf, kfull, Dv, modb, out);

    // LN + cm-modulate + MLP (WMMA x2) + gated residual, in place on d_out
    k_mlp<<<BB * (LL / 16), 256, 0, stream>>>(modb, w1b, b1, w2b, b2, out);
}
